// ComplexMultiHeadAttentionV2_74483322847295
// MI455X (gfx1250) — matrix operand order repaired and checked by reading, not compiled, not run on hardware
//
#include <hip/hip_runtime.h>

// ---------------------------------------------------------------------------
// ComplexMultiHeadAttention for MI455X (gfx1250, wave32, WMMA, async-LDS)
//   B=2, S=2048, D=1024, H=16, dk=64, dh=32
// Pipeline:
//   0) cvt_f32_f16     : hidden, W_q/k/v/o -> f16 (operands halved; L2-resident)
//   1) wmma_gemm_nt x3 : Q,K,V = hidden @ W.T           (4096x1024x1024, f16 in, f32 out)
//   2) feature_kernel  : euler -> qf/kf (b,h,s,64) f16, vT (b,h,d,s) f16
//   3) wmma_gemm_nt    : scores = qf @ kf^T             (per-head 2048x2048x64, causal skip)
//   4) softmax_kernel  : causal softmax in-place on attn region of d_out
//   5) wmma_gemm_nt    : ctx16 = attn @ V (via vT)      (per-head 2048x64x2048, f16 out)
//   6) wmma_gemm_nt    : out = ctx @ W_o.T              (4096x1024x1024)
// GEMM staging is double-buffered: GLOBAL_LOAD_ASYNC_TO_LDS_B128 issues tile
// i+1 while WMMAs consume tile i; "s_wait_asynccnt <= in-flight(i+1)" relies
// on in-order async-load completion to guarantee tile i has landed.
// ---------------------------------------------------------------------------

typedef __attribute__((ext_vector_type(16))) _Float16 v16h;
typedef __attribute__((ext_vector_type(8)))  _Float16 v8h;
typedef __attribute__((ext_vector_type(8)))  float    v8f;
typedef __attribute__((ext_vector_type(4)))  int      v4i;

typedef __attribute__((address_space(1))) v4i* g4ptr;  // global int4 vector ptr
typedef __attribute__((address_space(3))) v4i* l4ptr;  // LDS int4 vector ptr

#define LDK  40        // 32 K-halves + 8 pad: every 8-half group 16B aligned (row = 80B)
#define TILE (64 * LDK)

#if defined(__has_builtin)
#if __has_builtin(__builtin_amdgcn_global_load_async_to_lds_b128)
#define HAVE_ASYNC_LDS 1
#endif
#endif

template <int N>
__device__ __forceinline__ void wait_async() {
#if defined(__has_builtin)
#if __has_builtin(__builtin_amdgcn_s_wait_asynccnt)
  __builtin_amdgcn_s_wait_asynccnt(N);
  return;
#else
  asm volatile("s_wait_asynccnt %0" ::"n"(N) : "memory");
  return;
#endif
#else
  asm volatile("s_wait_asynccnt %0" ::"n"(N) : "memory");
#endif
}

// ---- LDS staging of one 64x32 f16 tile (128 threads) ----------------------
// f16 source: pure copy -> async-to-LDS (2 b128 issues per thread).
__device__ __forceinline__ void stage_tile(const _Float16* __restrict__ src,
                                           int row0, int lda, int k0,
                                           _Float16* dstLds, int t) {
#pragma unroll
  for (int i = 0; i < 2; ++i) {
    const int f  = t + i * 128;     // 0..255 b128 chunks
    const int r  = f >> 2;          // row 0..63
    const int c8 = (f & 3) * 8;     // half-col 0,8,16,24
    const _Float16* g = src + (long long)(row0 + r) * lda + k0 + c8;
    _Float16* l = &dstLds[r * LDK + c8];
#if HAVE_ASYNC_LDS
    __builtin_amdgcn_global_load_async_to_lds_b128(
        (g4ptr)(void*)g, (l4ptr)(void*)l, 0, 0);
#else
    *(v8h*)l = *(const v8h*)g;
#endif
  }
}

// f32 source: convert while staging (mandatory-f32 attn operand; ds-store path).
__device__ __forceinline__ void stage_tile(const float* __restrict__ src,
                                           int row0, int lda, int k0,
                                           _Float16* dstLds, int t) {
#pragma unroll
  for (int i = 0; i < 4; ++i) {
    const int f = t + i * 128;      // 0..511 float4 chunks
    const int r = f >> 3;           // row 0..63
    const int c = (f & 7) << 2;     // col 0,4,...,28
    float4 v = *(const float4*)(src + (long long)(row0 + r) * lda + k0 + c);
    _Float16* p = &dstLds[r * LDK + c];
    p[0] = (_Float16)v.x; p[1] = (_Float16)v.y;
    p[2] = (_Float16)v.z; p[3] = (_Float16)v.w;
  }
}

// A-fragment (16x32 f16): ISA layout  k(e) = 16*(e>>3) + 8*kh + (e&7)
__device__ __forceinline__ v16h frag_a(const _Float16* rowp, int kh) {
  v8h lo = *(const v8h*)(rowp + 8 * kh);
  v8h hi = *(const v8h*)(rowp + 16 + 8 * kh);
  v16h a;
#pragma unroll
  for (int i = 0; i < 8; ++i) { a[i] = lo[i]; a[i + 8] = hi[i]; }
  return a;
}

// B-fragment (32x16 f16): per lane n = lane&15, kh = lane>>4, k(e) = 16*(e>>3) + 8*kh + (e&7)
__device__ __forceinline__ v16h frag_b(const _Float16* rowp, int kh) {
  v8h lo = *(const v8h*)(rowp + 8 * kh);
  v8h hi = *(const v8h*)(rowp + 16 + 8 * kh);
  v16h b;
#pragma unroll
  for (int i = 0; i < 8; ++i) { b[i] = lo[i]; b[i + 8] = hi[i]; }
  return b;
}

// Generic batched NT GEMM: C[m,n] = sum_k A[m,k] * B[n,k]; B is f16,
// A is f16 or f32 (overloaded stager). Optional f32 and/or f16 C outputs.
// Batch z offsets: off = (z/zsplit)*s?1 + (z%zsplit)*s?2 (elements).
// Grid covers exactly: M,N % 64 == 0, K % 32 == 0 (>= 64).
template <typename TA>
__global__ __launch_bounds__(128)
void wmma_gemm_nt(const TA* __restrict__ A, const _Float16* __restrict__ B,
                  float* __restrict__ C, _Float16* __restrict__ C16,
                  int K, int lda, int ldb, int ldc,
                  long long sa1, long long sa2,
                  long long sb1, long long sb2,
                  long long sc1, long long sc2,
                  int zsplit, int causal) {
  const int bm = blockIdx.x, bn = blockIdx.y, z = blockIdx.z;
  if (causal && (bn * 64 > bm * 64 + 63)) return;  // tile fully above diagonal

  const TA*       Az = A + (long long)(z / zsplit) * sa1 + (long long)(z % zsplit) * sa2;
  const _Float16* Bz = B + (long long)(z / zsplit) * sb1 + (long long)(z % zsplit) * sb2;
  const long long coff = (long long)(z / zsplit) * sc1 + (long long)(z % zsplit) * sc2;

  __shared__ __align__(16) _Float16 As[2 * TILE];
  __shared__ __align__(16) _Float16 Bs[2 * TILE];

  const int t    = threadIdx.x;
  const int lane = t & 31;
  const int wave = t >> 5;
  const int wm   = wave >> 1;   // 2x2 wave grid over the 64x64 block tile
  const int wn   = wave & 1;
  const int lr   = lane & 15;
  const int kh   = lane >> 4;

  const v8f zero = {0.f, 0.f, 0.f, 0.f, 0.f, 0.f, 0.f, 0.f};
  v8f acc00 = zero, acc01 = zero, acc10 = zero, acc11 = zero;

  const int rowA0 = bm * 64;
  const int rowB0 = bn * 64;

  // Per-thread async issues per tile: f16 A -> 2 (A) + 2 (B); f32 A -> 2 (B only)
  constexpr int APT = (sizeof(TA) == 2) ? 4 : 2;

  // Software pipeline: tile 0 in flight before the loop.
  stage_tile(Az, rowA0, lda, 0, As, t);
  stage_tile(Bz, rowB0, ldb, 0, Bs, t);

  const int T = K >> 5;
  for (int i = 0; i < T; ++i) {
    _Float16* Ac = As + (i & 1) * TILE;
    _Float16* Bc = Bs + (i & 1) * TILE;
    if (i + 1 < T) {
      const int k0n = (i + 1) << 5;
      if (i + 2 < T) {  // L2 prefetch two tiles ahead
        __builtin_prefetch(Az + (long long)(rowA0 + (t >> 1)) * lda + k0n + 32, 0, 1);
        __builtin_prefetch(Bz + (long long)(rowB0 + (t >> 1)) * ldb + k0n + 32, 0, 1);
      }
      stage_tile(Az, rowA0, lda, k0n, As + ((i + 1) & 1) * TILE, t);
      stage_tile(Bz, rowB0, ldb, k0n, Bs + ((i + 1) & 1) * TILE, t);
#if HAVE_ASYNC_LDS
      wait_async<APT>();  // tile i landed; tile i+1 still streaming
#endif
    } else {
#if HAVE_ASYNC_LDS
      wait_async<0>();    // last tile: full drain
#endif
    }
    __syncthreads();

    v16h a0 = frag_a(&Ac[(wm * 32 +      lr) * LDK], kh);
    v16h a1 = frag_a(&Ac[(wm * 32 + 16 + lr) * LDK], kh);
    v16h b0 = frag_b(&Bc[(wn * 32 +      lr) * LDK], kh);
    v16h b1 = frag_b(&Bc[(wn * 32 + 16 + lr) * LDK], kh);

    acc00 = __builtin_amdgcn_wmma_f32_16x16x32_f16(false, a0, false, b0, (short)0, acc00, false, false);
    acc01 = __builtin_amdgcn_wmma_f32_16x16x32_f16(false, a0, false, b1, (short)0, acc01, false, false);
    acc10 = __builtin_amdgcn_wmma_f32_16x16x32_f16(false, a1, false, b0, (short)0, acc10, false, false);
    acc11 = __builtin_amdgcn_wmma_f32_16x16x32_f16(false, a1, false, b1, (short)0, acc11, false, false);
    __syncthreads();  // buffer i&1 free for rewrite at iteration i+1's stage
  }

  // C/D 16x16 f32 layout: VGPR v holds (m = v + 8*kh, n = lane&15)
  const int mb = bm * 64 + wm * 32 + kh * 8;
  const int nb = bn * 64 + wn * 32 + lr;
  v8f accs[4] = {acc00, acc01, acc10, acc11};
#pragma unroll
  for (int q = 0; q < 4; ++q) {
    const int mt = (q >> 1) * 16, nt = (q & 1) * 16;
    if (C) {
      float* p = C + coff + (long long)(mb + mt) * ldc + nb + nt;
#pragma unroll
      for (int v = 0; v < 8; ++v) { *p = accs[q][v]; p += ldc; }
    }
    if (C16) {
      _Float16* p = C16 + coff + (long long)(mb + mt) * ldc + nb + nt;
#pragma unroll
      for (int v = 0; v < 8; ++v) { *p = (_Float16)accs[q][v]; p += ldc; }
    }
  }
}

// f32 -> f16 bulk convert, 4 elements/thread (n % 4 == 0)
__global__ __launch_bounds__(256)
void cvt_f32_f16(const float* __restrict__ src, _Float16* __restrict__ dst,
                 long long n4) {
  long long i = ((long long)blockIdx.x * blockDim.x + threadIdx.x);
  if (i >= n4) return;
  float4 v = *(const float4*)(src + i * 4);
  _Float16* p = dst + i * 4;
  p[0] = (_Float16)v.x; p[1] = (_Float16)v.y;
  p[2] = (_Float16)v.z; p[3] = (_Float16)v.w;
}

// Euler decomposition + rotary-like phase features (f16 outputs).
// Q,K,V are (b,s,h,dk) f32; qf,kf -> (b,h,s,64); vT -> (b,h,dk,s).
__global__ __launch_bounds__(256)
void feature_kernel(const float* __restrict__ Q, const float* __restrict__ Kp,
                    const float* __restrict__ V,
                    const float* __restrict__ delta, const float* __restrict__ bias,
                    _Float16* __restrict__ qf, _Float16* __restrict__ kf,
                    _Float16* __restrict__ vT) {
  long long idx = (long long)blockIdx.x * blockDim.x + threadIdx.x;  // B*H*S*32
  const int j = (int)(idx & 31);
  const int s = (int)((idx >> 5) & 2047);
  const int h = (int)((idx >> 16) & 15);
  const int b = (int)(idx >> 20);

  const long long row = ((long long)(b * 2048 + s)) * 1024 + h * 64;  // (b,s,h,*)
  const float qr = Q[row + j],  qi = Q[row + 32 + j];
  const float kr = Kp[row + j], ki = Kp[row + 32 + j];

  const float mq = __fsqrt_rn(qr * qr + qi * qi + 1e-9f);
  const float pq = atan2f(qi, qr);
  const float mk = __fsqrt_rn(kr * kr + ki * ki + 1e-9f);
  const float pk = atan2f(ki, kr);

  const float freq = __powf(10000.0f, -(float)j * (1.0f / 32.0f));
  const float d  = delta[h * 32 + j];
  const float bb = bias[h * 32 + j];
  const float posf = (float)s * freq;
  const float Aang = d * pq + bb + posf;
  const float Bang = d * pk + posf;

  float sA, cA, sB, cB;
  __sincosf(Aang, &sA, &cA);
  __sincosf(Bang, &sB, &cB);

  const long long rf = (((long long)(b * 16 + h)) * 2048 + s) * 64;  // (b,h,s,*)
  qf[rf + j]      = (_Float16)(mq * cA);
  qf[rf + 32 + j] = (_Float16)(mq * sA);
  kf[rf + j]      = (_Float16)(mk * cB);
  kf[rf + 32 + j] = (_Float16)(mk * sB);

  const long long rv = (((long long)(b * 16 + h)) * 64 + j) * 2048 + s;  // (b,h,d,s)
  vT[rv]               = (_Float16)V[row + j];
  vT[rv + 32LL * 2048] = (_Float16)V[row + 32 + j];
}

// Causal row softmax, in place on the attn region of d_out.
__global__ __launch_bounds__(256)
void softmax_kernel(float* __restrict__ attn) {
  const int S = 2048;
  const long long row = blockIdx.x;        // B*H*S rows
  const int q = (int)(row & (S - 1));
  float* x = attn + row * (long long)S;
  const int L = q + 1;
  const int t = threadIdx.x;
  __shared__ float red[256];

  float m = -3.402823466e38f;
  for (int k = t; k < L; k += 256) m = fmaxf(m, x[k]);
  red[t] = m; __syncthreads();
  for (int off = 128; off > 0; off >>= 1) {
    if (t < off) red[t] = fmaxf(red[t], red[t + off]);
    __syncthreads();
  }
  const float mx = red[0];
  __syncthreads();

  float sum = 0.f;
  for (int k = t; k < L; k += 256) sum += __expf(x[k] - mx);
  red[t] = sum; __syncthreads();
  for (int off = 128; off > 0; off >>= 1) {
    if (t < off) red[t] += red[t + off];
    __syncthreads();
  }
  const float inv = 1.0f / red[0];
  __syncthreads();

  for (int k = t; k < L; k += 256) x[k] = __expf(x[k] - mx) * inv;
  for (int k = L + t; k < S; k += 256) x[k] = 0.0f;
}

extern "C" void kernel_launch(void* const* d_in, const int* in_sizes, int n_in,
                              void* d_out, int out_size, void* d_ws, size_t ws_size,
                              hipStream_t stream) {
  (void)in_sizes; (void)n_in; (void)out_size; (void)ws_size;
  const float* hidden = (const float*)d_in[0];
  // d_in[1] = attention_mask (causal tril) — implied analytically, not read
  const float* W_q  = (const float*)d_in[2];
  const float* W_k  = (const float*)d_in[3];
  const float* W_v  = (const float*)d_in[4];
  const float* W_o  = (const float*)d_in[5];
  const float* delt = (const float*)d_in[6];
  const float* bias = (const float*)d_in[7];

  const long long NE = 4194304LL;  // B*S*D
  const long long DD = 1048576LL;  // D*D
  char* w = (char*)d_ws;
  _Float16* h16   = (_Float16*)w; w += NE * 2;
  _Float16* Wq16  = (_Float16*)w; w += DD * 2;
  _Float16* Wk16  = (_Float16*)w; w += DD * 2;
  _Float16* Wv16  = (_Float16*)w; w += DD * 2;
  _Float16* Wo16  = (_Float16*)w; w += DD * 2;
  float*    Q     = (float*)w;    w += NE * 4;   // (b,s,h,d)
  float*    Kp    = (float*)w;    w += NE * 4;
  float*    V     = (float*)w;    w += NE * 4;
  _Float16* qf16  = (_Float16*)w; w += NE * 2;   // (b,h,s,64)
  _Float16* kf16  = (_Float16*)w; w += NE * 2;
  _Float16* vT16  = (_Float16*)w; w += NE * 2;   // (b,h,d,s)
  _Float16* ctx16 = (_Float16*)w; w += NE * 2;   // (b,s,h,d)

  float* out  = (float*)d_out;
  float* attn = out + NE;          // (b,h,q,k)

  const dim3 blk(128);
  const long long S = 2048, SS = S * S, SD = S * 64;

  // 0) f16 operand copies (L2-resident working set)
  cvt_f32_f16<<<(unsigned)(NE / 4 / 256), 256, 0, stream>>>(hidden, h16, NE / 4);
  cvt_f32_f16<<<(unsigned)(DD / 4 / 256), 256, 0, stream>>>(W_q, Wq16, DD / 4);
  cvt_f32_f16<<<(unsigned)(DD / 4 / 256), 256, 0, stream>>>(W_k, Wk16, DD / 4);
  cvt_f32_f16<<<(unsigned)(DD / 4 / 256), 256, 0, stream>>>(W_v, Wv16, DD / 4);
  cvt_f32_f16<<<(unsigned)(DD / 4 / 256), 256, 0, stream>>>(W_o, Wo16, DD / 4);

  // 1) Q/K/V projections
  {
    dim3 g(64, 16, 1);
    wmma_gemm_nt<_Float16><<<g, blk, 0, stream>>>(h16, Wq16, Q, (_Float16*)nullptr,
        1024, 1024, 1024, 1024, 0, 0, 0, 0, 0, 0, 1, 0);
    wmma_gemm_nt<_Float16><<<g, blk, 0, stream>>>(h16, Wk16, Kp, (_Float16*)nullptr,
        1024, 1024, 1024, 1024, 0, 0, 0, 0, 0, 0, 1, 0);
    wmma_gemm_nt<_Float16><<<g, blk, 0, stream>>>(h16, Wv16, V, (_Float16*)nullptr,
        1024, 1024, 1024, 1024, 0, 0, 0, 0, 0, 0, 1, 0);
  }

  // 2) Euler phase features + V transpose (f16 outputs)
  feature_kernel<<<8192, 256, 0, stream>>>(Q, Kp, V, delt, bias, qf16, kf16, vT16);

  // 3) scores = qf @ kf^T per (b,h), causal tiles skipped
  {
    dim3 g(32, 32, 32);
    wmma_gemm_nt<_Float16><<<g, blk, 0, stream>>>(qf16, kf16, attn, (_Float16*)nullptr,
        64, 64, 64, 2048, 16 * SD, SD, 16 * SD, SD, 16 * SS, SS, 16, 1);
  }

  // 4) causal softmax in-place (also zeroes masked region of attn output)
  softmax_kernel<<<65536, 256, 0, stream>>>(attn);

  // 5) ctx16 = attn @ V  (A is mandatory-f32 attn; B = vT16), ctx laid out (b,s,h,d)
  {
    dim3 g(32, 1, 32);
    wmma_gemm_nt<float><<<g, blk, 0, stream>>>(attn, vT16, (float*)nullptr, ctx16,
        2048, 2048, 2048, 1024, 16 * SS, SS, 16 * 64 * S, 64 * S,
        S * 1024, 64LL, 16, 0);
  }

  // 6) out = ctx @ W_o^T
  {
    dim3 g(64, 16, 1);
    wmma_gemm_nt<_Float16><<<g, blk, 0, stream>>>(ctx16, Wo16, out, (_Float16*)nullptr,
        1024, 1024, 1024, 1024, 0, 0, 0, 0, 0, 0, 1, 0);
  }
}
